// ConvLayer_81535659147296
// MI455X (gfx1250) — matrix operand order repaired and checked by reading, not compiled, not run on hardware
//
#include <hip/hip_runtime.h>
#include <hip/hip_bf16.h>

typedef __attribute__((ext_vector_type(16))) __bf16   v16bf;
typedef __attribute__((ext_vector_type(8)))  float    v8f;
typedef __attribute__((ext_vector_type(8)))  unsigned v8u;
typedef __attribute__((ext_vector_type(4)))  unsigned u32x4;
typedef __attribute__((ext_vector_type(8)))  int      i32x8;
typedef __attribute__((ext_vector_type(4)))  int      i32x4;

#define CIN   64
#define COUT  64
#define SLOPE 0.01f
#define EPSV  1e-5f

#if defined(__has_builtin)
#if __has_builtin(__builtin_amdgcn_tensor_load_to_lds) && \
    __has_builtin(__builtin_amdgcn_s_wait_tensorcnt)
#define USE_TDM 1
#endif
#endif

// ---------------------------------------------------------------------------
// Truncation-based fp32 -> bf16 hi/lo split, packed 2-at-a-time with v_perm.
// hi = trunc16(x) (residual exact in fp32); lo = trunc16(x - hi).
// Total representation error ~2^-16 relative -> 3-product WMMA gives ~fp32.
// Returns {hi_pack, lo_pack} (each holds 2 bf16: element0 = x0, element1 = x1).
// ---------------------------------------------------------------------------
__device__ __forceinline__ uint2 split_pair(float x0, float x1) {
    unsigned u0 = __float_as_uint(x0);
    unsigned u1 = __float_as_uint(x1);
    float l0 = x0 - __uint_as_float(u0 & 0xffff0000u);
    float l1 = x1 - __uint_as_float(u1 & 0xffff0000u);
    uint2 r;
    r.x = __builtin_amdgcn_perm(u1, u0, 0x07060302u);   // {u1.hi16, u0.hi16}
    r.y = __builtin_amdgcn_perm(__float_as_uint(l1), __float_as_uint(l0),
                                0x07060302u);
    return r;
}

// A fragment (16x32 bf16 layout): lane holds row (lane&15); for lane-half h:
// element e<8 -> K = chunk*32 + 8h + e ; e>=8 -> K = chunk*32 + 8h + 16 + (e-8)
__device__ __forceinline__ void load_a_frag(const float* __restrict__ row,
                                            int chunk, int lhalf,
                                            v16bf& hi, v16bf& lo) {
    const float4* p = (const float4*)(row + chunk * 32 + 8 * lhalf);
    float4 q0 = p[0];   // elements 0-3
    float4 q1 = p[1];   // elements 4-7
    float4 q2 = p[4];   // elements 8-11  (K offset +16)
    float4 q3 = p[5];   // elements 12-15 (K offset +20)
    v8u H, L;
    uint2 r;
    r = split_pair(q0.x, q0.y); H[0] = r.x; L[0] = r.y;
    r = split_pair(q0.z, q0.w); H[1] = r.x; L[1] = r.y;
    r = split_pair(q1.x, q1.y); H[2] = r.x; L[2] = r.y;
    r = split_pair(q1.z, q1.w); H[3] = r.x; L[3] = r.y;
    r = split_pair(q2.x, q2.y); H[4] = r.x; L[4] = r.y;
    r = split_pair(q2.z, q2.w); H[5] = r.x; L[5] = r.y;
    r = split_pair(q3.x, q3.y); H[6] = r.x; L[6] = r.y;
    r = split_pair(q3.z, q3.w); H[7] = r.x; L[7] = r.y;
    hi = __builtin_bit_cast(v16bf, H);
    lo = __builtin_bit_cast(v16bf, L);
}

__device__ __forceinline__ v8f wmma_bf16(v16bf a, v16bf b, v8f c) {
    return __builtin_amdgcn_wmma_f32_16x16x32_bf16(
        false, a, false, b, (short)0, c, false, false);
}

// ---------------------------------------------------------------------------
// Stage one 64x64 fp32 weight tile into LDS.
// Preferred path: Tensor Data Mover (TDM) async DMA + s_wait_tensorcnt.
// D# per CDNA5 ISA sec.8: 2-D tensor, data_size=4B, tile 64x64.
// ---------------------------------------------------------------------------
__device__ __forceinline__ void stage_w_to_lds(const float* __restrict__ gsrc,
                                               float* ldsDst) {
#ifdef USE_TDM
    if (threadIdx.x < 32) {                 // one wave issues the DMA
        unsigned long long ga = (unsigned long long)(size_t)gsrc;
        unsigned ldsOff = (unsigned)(size_t)ldsDst;   // low 32b = LDS offset
        u32x4 g0;
        g0[0] = 1u;                                   // count=1, user mode
        g0[1] = ldsOff;                               // lds_addr
        g0[2] = (unsigned)(ga & 0xffffffffu);         // global_addr[31:0]
        g0[3] = (unsigned)((ga >> 32) & 0x01ffffffu)  // global_addr[56:32]
              | (2u << 30);                           // type = 2 ("image")
        i32x8 g1;
        g1[0] = 0x00020000;        // workgroup_mask=0, data_size=2 (4 bytes)
        g1[1] = (int)(64u << 16);  // tensor_dim0[15:0]=64 in bits[63:48]
        g1[2] = (int)(64u << 16);  // tensor_dim1[15:0]=64 in bits[95:80]
        g1[3] = (int)(64u << 16);  // tile_dim0=64 in bits[127:112]
        g1[4] = 64;                // tile_dim1=64, tile_dim2=0
        g1[5] = 64;                // tensor_dim0_stride[31:0]=64
        g1[6] = 0;
        g1[7] = 0;
        i32x4 gz = {0, 0, 0, 0};
#if __clang_major__ >= 23
        i32x8 gz8 = {0, 0, 0, 0, 0, 0, 0, 0};
        __builtin_amdgcn_tensor_load_to_lds(g0, g1, gz, gz, gz8, 0);
#else
        __builtin_amdgcn_tensor_load_to_lds(g0, g1, gz, gz, 0);
#endif
        __builtin_amdgcn_s_wait_tensorcnt(0);
    }
    __syncthreads();
#else
    for (int i = threadIdx.x; i < CIN * COUT; i += blockDim.x)
        ldsDst[i] = gsrc[i];
    __syncthreads();
#endif
}

// Convert staged fp32 W into bf16 hi/lo LDS arrays (column-readable for B).
__device__ __forceinline__ void convert_w(const float* sWstage,
                                          __bf16* sWhi, __bf16* sWlo) {
    for (int i = threadIdx.x; i < CIN * COUT; i += blockDim.x) {
        float w = sWstage[i];
        unsigned u = __float_as_uint(w);
        unsigned hu = u & 0xffff0000u;
        float lo = w - __uint_as_float(hu);
        sWhi[i] = __builtin_bit_cast(__bf16, (unsigned short)(hu >> 16));
        sWlo[i] = __builtin_bit_cast(__bf16,
                      (unsigned short)(__float_as_uint(lo) >> 16));
    }
    __syncthreads();
}

// Build B fragments for all 4 column tiles x 2 K-chunks (held in registers).
// B (32x16 bf16) layout: lane = column, element e -> K = 8*(lane>>4) + e (e<8), 8*(lane>>4) + 16 + (e-8) (e>=8).
__device__ __forceinline__ void build_b_frags(const __bf16* sWhi,
                                              const __bf16* sWlo,
                                              int lane, int lhalf,
                                              v16bf bhi[4][2], v16bf blo[4][2]) {
#pragma unroll
    for (int nt = 0; nt < 4; ++nt) {
        const int col = (lane & 15) + 16 * nt;
#pragma unroll
        for (int c = 0; c < 2; ++c) {
            const int kbase = c * 32 + 8 * lhalf;
#pragma unroll
            for (int e = 0; e < 16; ++e) {
                bhi[nt][c][e] = sWhi[(kbase + e + (e >= 8 ? 8 : 0)) * COUT + col];
                blo[nt][c][e] = sWlo[(kbase + e + (e >= 8 ? 8 : 0)) * COUT + col];
            }
        }
    }
}

// ---------------- kernel 0: zero the stats scratch -------------------------
__global__ void zero_stats_kernel(float* stats) {
    if (threadIdx.x < 128) stats[threadIdx.x] = 0.0f;
}

// ---------------- kernel 1: center tap (dense GEMM into d_out) -------------
// 256 threads = 8 waves; each wave independently owns 16-row tiles and
// computes the full 16x64 output (all 4 column tiles).
__global__ void __launch_bounds__(256)
center_gemm_kernel(const float* __restrict__ feat,
                   const float* __restrict__ W,
                   float* __restrict__ out,
                   int N, int numTiles) {
    __shared__ float  sWstage[CIN * COUT];
    __shared__ __bf16 sWhi[CIN * COUT];
    __shared__ __bf16 sWlo[CIN * COUT];

    stage_w_to_lds(W + 13 * CIN * COUT, sWstage);
    convert_w(sWstage, sWhi, sWlo);

    const int wave  = threadIdx.x >> 5;
    const int lane  = threadIdx.x & 31;
    const int lhalf = lane >> 4;

    v16bf bhi[4][2], blo[4][2];
    build_b_frags(sWhi, sWlo, lane, lhalf, bhi, blo);

    const int waveStride = gridDim.x * 8;
    for (int tile = blockIdx.x * 8 + wave; tile < numTiles; tile += waveStride) {
        const int r = tile * 16 + (lane & 15);
        const float* rowp = feat + (size_t)(r < N ? r : 0) * CIN;

        v16bf ahi[2], alo[2];
        load_a_frag(rowp, 0, lhalf, ahi[0], alo[0]);
        load_a_frag(rowp, 1, lhalf, ahi[1], alo[1]);

        const int rbase = tile * 16 + 8 * lhalf;
#pragma unroll
        for (int nt = 0; nt < 4; ++nt) {
            v8f acc = {};
#pragma unroll
            for (int c = 0; c < 2; ++c) {
                acc = wmma_bf16(ahi[c], bhi[nt][c], acc);
                acc = wmma_bf16(ahi[c], blo[nt][c], acc);
                acc = wmma_bf16(alo[c], bhi[nt][c], acc);
            }
            const int col = (lane & 15) + 16 * nt;
#pragma unroll
            for (int v = 0; v < 8; ++v) {
                const int rr = rbase + v;
                if (rr < N) out[(size_t)rr * COUT + col] = acc[v];
            }
        }
    }
}

// ---------------- kernel 2: 26 neighbor taps, gather-GEMM-scatter ----------
__global__ void __launch_bounds__(256)
neighbor_gemm_kernel(const float* __restrict__ feat,
                     const float* __restrict__ W,
                     const int* __restrict__ in_idx,   // (26,M)
                     const int* __restrict__ out_idx,  // (26,M), ==N for pad
                     float* __restrict__ out,
                     int N, int M, int bpk) {
    __shared__ float  sWstage[CIN * COUT];
    __shared__ __bf16 sWhi[CIN * COUT];
    __shared__ __bf16 sWlo[CIN * COUT];

    const int k    = blockIdx.x / bpk;       // neighbor offset 0..25
    const int widx = (k < 13) ? k : k + 1;   // skip center weight

    stage_w_to_lds(W + (size_t)widx * CIN * COUT, sWstage);
    convert_w(sWstage, sWhi, sWlo);

    const int wave  = threadIdx.x >> 5;
    const int lane  = threadIdx.x & 31;
    const int lhalf = lane >> 4;

    v16bf bhi[4][2], blo[4][2];
    build_b_frags(sWhi, sWlo, lane, lhalf, bhi, blo);

    const int* iidx = in_idx  + (size_t)k * M;
    const int* oidx = out_idx + (size_t)k * M;
    const int  numTiles   = M >> 4;
    const int  waveStride = bpk * 8;

    for (int tile = (blockIdx.x % bpk) * 8 + wave; tile < numTiles;
         tile += waveStride) {
        const int m = tile * 16 + (lane & 15);
        const int r = iidx[m];                      // gather row (0 for pad)
        const float* rowp = feat + (size_t)r * CIN;

        v16bf ahi[2], alo[2];
        load_a_frag(rowp, 0, lhalf, ahi[0], alo[0]);
        load_a_frag(rowp, 1, lhalf, ahi[1], alo[1]);

        const int mbase = tile * 16 + 8 * lhalf;
        int orow[8];
#pragma unroll
        for (int v = 0; v < 8; ++v) orow[v] = oidx[mbase + v];

#pragma unroll
        for (int nt = 0; nt < 4; ++nt) {
            v8f acc = {};
#pragma unroll
            for (int c = 0; c < 2; ++c) {
                acc = wmma_bf16(ahi[c], bhi[nt][c], acc);
                acc = wmma_bf16(ahi[c], blo[nt][c], acc);
                acc = wmma_bf16(alo[c], bhi[nt][c], acc);
            }
            const int col = (lane & 15) + 16 * nt;
#pragma unroll
            for (int v = 0; v < 8; ++v) {
                if (orow[v] < N)                     // N == padding sentinel
                    atomicAdd(&out[(size_t)orow[v] * COUT + col], acc[v]);
            }
        }
    }
}

// ---------------- kernel 3: per-channel sum / sumsq of leaky(conv) ---------
__global__ void __launch_bounds__(256)
leaky_stats_kernel(const float* __restrict__ conv, float* __restrict__ stats,
                   int total) {
    __shared__ float ls[256];
    __shared__ float ls2[256];
    float s = 0.0f, s2 = 0.0f;
    const int stride = gridDim.x * blockDim.x;   // multiple of 64
    for (int i = blockIdx.x * blockDim.x + threadIdx.x; i < total; i += stride) {
        float v = conv[i];
        float a = v >= 0.0f ? v : SLOPE * v;
        s  += a;
        s2 += a * a;
    }
    ls[threadIdx.x]  = s;
    ls2[threadIdx.x] = s2;
    __syncthreads();
    if (threadIdx.x < 64) {
        float ts  = ls[threadIdx.x]  + ls[threadIdx.x + 64]
                  + ls[threadIdx.x + 128] + ls[threadIdx.x + 192];
        float ts2 = ls2[threadIdx.x] + ls2[threadIdx.x + 64]
                  + ls2[threadIdx.x + 128] + ls2[threadIdx.x + 192];
        atomicAdd(&stats[threadIdx.x],      ts);
        atomicAdd(&stats[64 + threadIdx.x], ts2);
    }
}

// ---------------- kernel 4: fused leaky + batch-norm affine ----------------
__global__ void __launch_bounds__(256)
normalize_kernel(float* __restrict__ out, const float* __restrict__ stats,
                 const float* __restrict__ gamma, const float* __restrict__ beta,
                 float invN, int total) {
    const int ch = threadIdx.x & 63;             // invariant per thread
    const float mean  = stats[ch] * invN;
    const float var   = stats[64 + ch] * invN - mean * mean;
    const float scale = gamma[ch] * rsqrtf(var + EPSV);
    const float bias  = beta[ch] - mean * scale;
    const int stride = gridDim.x * blockDim.x;
    for (int i = blockIdx.x * blockDim.x + threadIdx.x; i < total; i += stride) {
        float v = out[i];
        float a = v >= 0.0f ? v : SLOPE * v;     // recompute leaky in place
        out[i] = a * scale + bias;
    }
}

// ------------------------------ launcher -----------------------------------
extern "C" void kernel_launch(void* const* d_in, const int* in_sizes, int n_in,
                              void* d_out, int out_size, void* d_ws, size_t ws_size,
                              hipStream_t stream) {
    const float* feat    = (const float*)d_in[0];   // (N,64)
    const float* W       = (const float*)d_in[1];   // (27,64,64)
    const float* gamma   = (const float*)d_in[2];   // (64,)
    const float* beta    = (const float*)d_in[3];   // (64,)
    const int*   in_idx  = (const int*)d_in[4];     // (26,M)
    const int*   out_idx = (const int*)d_in[5];     // (26,M)
    float* out   = (float*)d_out;                   // (N,64)
    float* stats = (float*)d_ws;                    // 128 floats: sum | sumsq

    const int N = in_sizes[0] / CIN;
    const int M = in_sizes[4] / 26;
    const int total = N * COUT;
    const int numTiles = (N + 15) / 16;

    zero_stats_kernel<<<1, 128, 0, stream>>>(stats);

    center_gemm_kernel<<<512, 256, 0, stream>>>(feat, W, out, N, numTiles);

    const int bpk = 32;                             // blocks per neighbor offset
    neighbor_gemm_kernel<<<26 * bpk, 256, 0, stream>>>(feat, W, in_idx, out_idx,
                                                       out, N, M, bpk);

    leaky_stats_kernel<<<512, 256, 0, stream>>>(out, stats, total);

    normalize_kernel<<<512, 256, 0, stream>>>(out, stats, gamma, beta,
                                              1.0f / (float)N, total);
}
